// Model_52518860096214
// MI455X (gfx1250) — matrix operand order repaired and checked by reading, not compiled, not run on hardware
//
#include <hip/hip_runtime.h>
#include <math.h>

// ---------------------------------------------------------------------------
// Attention block for MI455X (gfx1250, wave32):
//   qkv = x @ qkv_w^T + qkv_b ; scores = (q k^T)/sqrt(768) ; softmax ;
//   ctx = attn @ v ; out = ctx @ out_w^T + out_b
// GEMMs: v_wmma_f32_16x16x32_bf16, float4 global loads, v_perm_b32 packed
// f32->bf16 conversion, b64/b32 LDS stores, ping-pong LDS double buffering.
// ---------------------------------------------------------------------------

typedef __attribute__((ext_vector_type(16))) __bf16 v16bf;
typedef __attribute__((ext_vector_type(8)))  float  v8f;

#define DIMK 768
#define NBAT 4
#define SEQ  2048

#define TM 128   // block tile M
#define TN 128   // block tile N
#define TK 32    // K chunk (one bf16 WMMA K)
#define LROW 40  // padded LDS row in bf16 elems (32 + 8) -> 80B rows

// round-to-nearest (ties away) f32 -> bf16; 2 ops
__device__ __forceinline__ unsigned short f2bf(float f) {
    unsigned int u = __builtin_bit_cast(unsigned int, f) + 0x8000u;
    return (unsigned short)(u >> 16);
}

// pack bf16(a) into [15:0], bf16(b) into [31:16]: 2 adds + 1 v_perm_b32
__device__ __forceinline__ unsigned int pack2(float a, float b) {
    unsigned int ua = __builtin_bit_cast(unsigned int, a) + 0x8000u;
    unsigned int ub = __builtin_bit_cast(unsigned int, b) + 0x8000u;
    // bytes: [3]=ub.b3 [2]=ub.b2 [1]=ua.b3 [0]=ua.b2  (src0=ub -> sel 4..7)
    return __builtin_amdgcn_perm(ub, ua, 0x07060302u);
}

__device__ __forceinline__ void store_tile(float* __restrict__ Cb, long ldc,
                                           int rowb, int col,
                                           const float* __restrict__ bias,
                                           float scale, v8f a, int g) {
    float badd = bias ? bias[col] : 0.0f;
#pragma unroll
    for (int r = 0; r < 8; ++r)
        Cb[(long)(rowb + r + 8 * g) * ldc + col] = a[r] * scale + badd;
}

// C[m,n] = scale * sum_k A[m*lda + k] * B[n*bn + k*bk] + bias[n]
// Batched via blockIdx.z with per-operand batch strides.
__global__ __launch_bounds__(256)
void gemm_bt_bf16(const float* __restrict__ A,
                  const float* __restrict__ Bm,
                  const float* __restrict__ bias,
                  float* __restrict__ C,
                  int K,
                  long lda, long bn, long bk, long ldc,
                  long abs_, long bbs_, long cbs_,
                  float scale, int bKmajor)
{
    __shared__ unsigned short sA[2][TM][LROW];
    __shared__ unsigned short sB[2][TN][LROW];

    const int tid  = threadIdx.x;
    const int lane = tid & 31;
    const int wid  = tid >> 5;
    const int l16  = lane & 15;
    const int g    = lane >> 4;

    const int m0 = blockIdx.y * TM;
    const int n0 = blockIdx.x * TN;

    const float* Ab = A  + (long)blockIdx.z * abs_;
    const float* Bb = Bm + (long)blockIdx.z * bbs_;
    float*       Cb = C  + (long)blockIdx.z * cbs_;

    const int wm = (wid & 3) * 32;   // wave's 32 rows inside block tile
    const int wn = (wid >> 2) * 64;  // wave's 64 cols inside block tile

    // ---- per-thread staging coordinates & flat offsets (hoisted) ----
    // A (and B when k-contiguous): 4 items, r = idx>>3 in [0,128), c4 = idx&7
    int arr[4], ac4[4];
    long offA[4];
#pragma unroll
    for (int i = 0; i < 4; ++i) {
        int idx = i * 256 + tid;
        arr[i] = idx >> 3;
        ac4[i] = idx & 7;
        offA[i] = (long)(m0 + arr[i]) * lda + ac4[i] * 4;
    }
    // B offsets (two layouts)
    long offB[4];
    int bn4[2], bkk[2];
    if (bKmajor) {
        // 2 items, each covers rows k=2*kkk and 2*kkk+1, 4 consecutive n
#pragma unroll
        for (int i = 0; i < 2; ++i) {
            int idx = i * 256 + tid;
            bkk[i] = (idx >> 5) * 2;          // even k in [0,32)
            bn4[i] = idx & 31;                // n4 in [0,32)
            offB[2 * i]     = (long)bkk[i] * bk + (n0 + bn4[i] * 4);
            offB[2 * i + 1] = offB[2 * i] + bk;
        }
    } else {
#pragma unroll
        for (int i = 0; i < 4; ++i)
            offB[i] = (long)(n0 + arr[i]) * bn + ac4[i] * 4;
    }

    v8f acc[2][4];
#pragma unroll
    for (int ti = 0; ti < 2; ++ti)
#pragma unroll
        for (int tj = 0; tj < 4; ++tj)
            acc[ti][tj] = (v8f){};

    // ---- prologue: stage chunk 0 into buffer 0 ----
    {
#pragma unroll
        for (int i = 0; i < 4; ++i) {
            float4 va = *reinterpret_cast<const float4*>(&Ab[offA[i]]);
            *reinterpret_cast<uint2*>(&sA[0][arr[i]][ac4[i] * 4]) =
                make_uint2(pack2(va.x, va.y), pack2(va.z, va.w));
        }
        if (bKmajor) {
#pragma unroll
            for (int i = 0; i < 2; ++i) {
                float4 lo = *reinterpret_cast<const float4*>(&Bb[offB[2 * i]]);
                float4 hi = *reinterpret_cast<const float4*>(&Bb[offB[2 * i + 1]]);
                int nb0 = bn4[i] * 4, kk = bkk[i];
                *(unsigned int*)&sB[0][nb0 + 0][kk] = pack2(lo.x, hi.x);
                *(unsigned int*)&sB[0][nb0 + 1][kk] = pack2(lo.y, hi.y);
                *(unsigned int*)&sB[0][nb0 + 2][kk] = pack2(lo.z, hi.z);
                *(unsigned int*)&sB[0][nb0 + 3][kk] = pack2(lo.w, hi.w);
            }
        } else {
#pragma unroll
            for (int i = 0; i < 4; ++i) {
                float4 vb = *reinterpret_cast<const float4*>(&Bb[offB[i]]);
                *reinterpret_cast<uint2*>(&sB[0][arr[i]][ac4[i] * 4]) =
                    make_uint2(pack2(vb.x, vb.y), pack2(vb.z, vb.w));
            }
        }
    }

    int buf = 0;
    for (int kb = 0; kb < K; kb += TK) {
        __syncthreads();

        const bool hasNext = (kb + TK) < K;
        float4 nA[4], nB[4];

        // ---- issue next chunk's global loads early (hide latency) ----
        if (hasNext) {
            const int kn = kb + TK;
#pragma unroll
            for (int i = 0; i < 4; ++i)
                nA[i] = *reinterpret_cast<const float4*>(&Ab[offA[i] + kn]);
            if (bKmajor) {
                const long kofs = (long)kn * bk;
#pragma unroll
                for (int i = 0; i < 4; ++i)
                    nB[i] = *reinterpret_cast<const float4*>(&Bb[offB[i] + kofs]);
            } else {
#pragma unroll
                for (int i = 0; i < 4; ++i)
                    nB[i] = *reinterpret_cast<const float4*>(&Bb[offB[i] + kn]);
            }
            if (kb + 2 * TK < K)   // engage global_prefetch path for chunk i+2
                __builtin_prefetch(&Ab[offA[0] + kb + 2 * TK], 0, 1);
        }

        // ---- fragments per CDNA5 WMMA layouts, from current buffer ----
        union Frag { v16bf v; unsigned int u[8]; };
        Frag fa[2], fb[4];
#pragma unroll
        for (int v = 0; v < 8; ++v) {
            // A 16x32: VGPR v holds a K pair; lanes 16-31 use K+8 per half
            int ka = (v < 4) ? (2 * v + 8 * g) : (16 + 2 * (v - 4) + 8 * g);
#pragma unroll
            for (int ti = 0; ti < 2; ++ti)
                fa[ti].u[v] = *(const unsigned int*)&sA[buf][wm + ti * 16 + l16][ka];
            // B 32x16: N = lane%16; same K pairs as A (lanes 16-31 use K+8 per half)
            int kbv = ka;
#pragma unroll
            for (int tj = 0; tj < 4; ++tj)
                fb[tj].u[v] = *(const unsigned int*)&sB[buf][wn + tj * 16 + l16][kbv];
        }

#pragma unroll
        for (int ti = 0; ti < 2; ++ti)
#pragma unroll
            for (int tj = 0; tj < 4; ++tj)
                acc[ti][tj] = __builtin_amdgcn_wmma_f32_16x16x32_bf16(
                    false, fa[ti].v, false, fb[tj].v,
                    (short)0, acc[ti][tj], false, false);

        // ---- store staged next chunk into the other buffer ----
        if (hasNext) {
            const int nb = buf ^ 1;
#pragma unroll
            for (int i = 0; i < 4; ++i)
                *reinterpret_cast<uint2*>(&sA[nb][arr[i]][ac4[i] * 4]) =
                    make_uint2(pack2(nA[i].x, nA[i].y), pack2(nA[i].z, nA[i].w));
            if (bKmajor) {
#pragma unroll
                for (int i = 0; i < 2; ++i) {
                    float4 lo = nB[2 * i], hi = nB[2 * i + 1];
                    int nb0 = bn4[i] * 4, kk = bkk[i];
                    *(unsigned int*)&sB[nb][nb0 + 0][kk] = pack2(lo.x, hi.x);
                    *(unsigned int*)&sB[nb][nb0 + 1][kk] = pack2(lo.y, hi.y);
                    *(unsigned int*)&sB[nb][nb0 + 2][kk] = pack2(lo.z, hi.z);
                    *(unsigned int*)&sB[nb][nb0 + 3][kk] = pack2(lo.w, hi.w);
                }
            } else {
#pragma unroll
                for (int i = 0; i < 4; ++i)
                    *reinterpret_cast<uint2*>(&sB[nb][arr[i]][ac4[i] * 4]) =
                        make_uint2(pack2(nB[i].x, nB[i].y), pack2(nB[i].z, nB[i].w));
            }
        }
        buf ^= 1;
    }

    // ---- epilogue: C layout -> lane = N, VGPR r -> M = r + 8*(lane/16) ----
#pragma unroll
    for (int ti = 0; ti < 2; ++ti)
#pragma unroll
        for (int tj = 0; tj < 4; ++tj)
            store_tile(Cb, ldc, m0 + wm + ti * 16, n0 + wn + tj * 16 + l16,
                       bias, scale, acc[ti][tj], g);
}

// One block (8 waves) per row; ncols = 2048.
__global__ __launch_bounds__(256)
void softmax_rows(float* __restrict__ data, int ncols)
{
    __shared__ float red[8];
    float* p = data + (long)blockIdx.x * ncols;
    const int tid = threadIdx.x;
    const int lane = tid & 31, wid = tid >> 5;

    float m = -3.4e38f;
    for (int i = tid; i < ncols; i += 256) m = fmaxf(m, p[i]);
#pragma unroll
    for (int o = 16; o > 0; o >>= 1) m = fmaxf(m, __shfl_xor(m, o, 32));
    if (lane == 0) red[wid] = m;
    __syncthreads();
    float rm = red[0];
#pragma unroll
    for (int w = 1; w < 8; ++w) rm = fmaxf(rm, red[w]);
    __syncthreads();

    float s = 0.0f;
    for (int i = tid; i < ncols; i += 256) {
        float e = __expf(p[i] - rm);
        p[i] = e;
        s += e;
    }
#pragma unroll
    for (int o = 16; o > 0; o >>= 1) s += __shfl_xor(s, o, 32);
    if (lane == 0) red[wid] = s;
    __syncthreads();
    float rs = 0.0f;
#pragma unroll
    for (int w = 0; w < 8; ++w) rs += red[w];
    float inv = 1.0f / rs;
    for (int i = tid; i < ncols; i += 256) p[i] *= inv;
}

extern "C" void kernel_launch(void* const* d_in, const int* in_sizes, int n_in,
                              void* d_out, int out_size, void* d_ws, size_t ws_size,
                              hipStream_t stream)
{
    (void)in_sizes; (void)n_in; (void)out_size; (void)ws_size;
    const float* x     = (const float*)d_in[0];   // [4,2048,768]
    const float* qkv_w = (const float*)d_in[1];   // [2304,768]
    const float* qkv_b = (const float*)d_in[2];   // [2304]
    const float* out_w = (const float*)d_in[3];   // [768,768]
    const float* out_b = (const float*)d_in[4];   // [768]
    float* out = (float*)d_out;                   // [4,2048,768]

    const long Mtok = (long)NBAT * SEQ;           // 8192
    float* qkv    = (float*)d_ws;                 // 8192 x 2304
    float* scores = qkv + Mtok * (3 * DIMK);      // 4 x 2048 x 2048
    float* ctx    = scores + (long)NBAT * SEQ * SEQ;  // 8192 x 768

    dim3 blk(256);

    // 1) qkv = x @ qkv_w^T + qkv_b
    gemm_bt_bf16<<<dim3((3 * DIMK) / TN, Mtok / TM, 1), blk, 0, stream>>>(
        x, qkv_w, qkv_b, qkv,
        DIMK, DIMK, DIMK, 1, 3 * DIMK,
        0, 0, 0, 1.0f, 0);

    // 2) scores = (q @ k^T) / sqrt(768)   (q scaling folded into scale)
    gemm_bt_bf16<<<dim3(SEQ / TN, SEQ / TM, NBAT), blk, 0, stream>>>(
        qkv, qkv + DIMK, nullptr, scores,
        DIMK, 3 * DIMK, 3 * DIMK, 1, SEQ,
        (long)SEQ * 3 * DIMK, (long)SEQ * 3 * DIMK, (long)SEQ * SEQ,
        1.0f / sqrtf((float)DIMK), 0);

    // 3) softmax over each of the 8192 rows of length 2048
    softmax_rows<<<dim3(NBAT * SEQ), blk, 0, stream>>>(scores, SEQ);

    // 4) ctx = attn @ v   (V accessed K-major: bn=1, bk=3*DIM)
    gemm_bt_bf16<<<dim3(DIMK / TN, SEQ / TM, NBAT), blk, 0, stream>>>(
        scores, qkv + 2 * DIMK, nullptr, ctx,
        SEQ, SEQ, 1, 3 * DIMK, DIMK,
        (long)SEQ * SEQ, (long)SEQ * 3 * DIMK, (long)SEQ * DIMK,
        1.0f, 1);

    // 5) out = ctx @ out_w^T + out_b
    gemm_bt_bf16<<<dim3(DIMK / TN, Mtok / TM, 1), blk, 0, stream>>>(
        ctx, out_w, out_b, out,
        DIMK, DIMK, DIMK, 1, DIMK,
        0, 0, 0, 1.0f, 0);
}
